// CrossViewAttention_60619168416524
// MI455X (gfx1250) — matrix operand order repaired and checked by reading, not compiled, not run on hardware
//
#include <hip/hip_runtime.h>
#include <hip/hip_bf16.h>

typedef __attribute__((ext_vector_type(16))) __bf16 v16bf;
typedef __attribute__((ext_vector_type(8)))  float  v8f;
typedef unsigned short u16;
typedef unsigned int   u32;

#define BATCH 8
#define CCH   512   // channels
#define BN    4096  // N = H*W
#define DQK   64    // C/8

#if __has_builtin(__builtin_amdgcn_global_load_async_to_lds_b128) && \
    __has_builtin(__builtin_amdgcn_s_wait_asynccnt)
#define USE_ASYNC_LDS 1
typedef __attribute__((ext_vector_type(4))) int i32x4;
typedef __attribute__((address_space(1))) i32x4 gi32x4;   // global int4
typedef __attribute__((address_space(3))) i32x4 si32x4;   // LDS int4
#else
#define USE_ASYNC_LDS 0
#endif

// ---------- helpers ----------

__device__ inline u16 f2bf(float x) {   // hardware f32 -> bf16 convert
  const __bf16 h = (__bf16)x;
  return __builtin_bit_cast(u16, h);
}

#if USE_ASYNC_LDS
// one 16-byte async global->LDS transfer per active lane (ASYNCcnt-tracked)
__device__ inline void async_copy16(const u16* gsrc, u16* ldst) {
  __builtin_amdgcn_global_load_async_to_lds_b128(
      __builtin_bit_cast(gi32x4*, gsrc), (si32x4*)ldst, 0, 0);
}
#endif

union Frag16 { v16bf v; u16 h[16]; u32 w[8]; };

// A-operand (16x32 bf16). Logical element (m,k) at base[m*ld + k] (k contiguous).
// ISA layout: lanes 0-15 row M=lane, V0-3 K=0..7, V4-7 K=16..23; lanes 16-31 K+=8.
__device__ inline v16bf load_fragA_u16(const u16* base, int ld) {
  const int lane = threadIdx.x & 31;
  const int m = lane & 15, hi = lane >> 4;
  Frag16 f;
#pragma unroll
  for (int i = 0; i < 8; ++i) {
    const int k = ((i & 4) << 2) + hi * 8 + (i & 3) * 2;  // (i<4?0:16)+hi*8+(i%4)*2
    f.w[i] = *(const u32*)(base + m * ld + k);
  }
  return f.v;
}

// B-operand (32x16 bf16) stored transposed: element (k,n) at base[n*ld + k].
// ISA layout: lanes 0-15 col N=lane, V0-3 K=0..7, V4-7 K=16..23; lanes 16-31 K+=8.
__device__ inline v16bf load_fragB_u16(const u16* base, int ld) {
  const int lane = threadIdx.x & 31;
  const int n = lane & 15, hi = lane >> 4;
  Frag16 f;
#pragma unroll
  for (int i = 0; i < 8; ++i) {
    const int k = ((i & 4) << 2) + hi * 8 + (i & 3) * 2;
    f.w[i] = *(const u32*)(base + n * ld + k);
  }
  return f.v;
}

__device__ inline v8f wmma_bf16(v16bf a, v16bf b, v8f c) {
  return __builtin_amdgcn_wmma_f32_16x16x32_bf16(false, a, false, b, (short)0, c,
                                                 false, false);
}

// ---------- one-shot fp32 -> bf16 weight conversion ----------
__global__ __launch_bounds__(256) void cvt_bf16_kernel(const float* __restrict__ src,
                                                       u16* __restrict__ dst, int n) {
  const int i = blockIdx.x * 256 + threadIdx.x;
  if (i < n) dst[i] = f2bf(src[i]);
}

// ---------- pass A: q/k projection  out[b,n,o] = sum_c view[b,c,n]*W[o,c] + bias[o] ----------
// grid (N/64, 1, B), 128 threads (4 waves). Wave w owns rows n = 16w..16w+15, all 64 outputs.
// Wb = bf16 weights, row-major [o][c].
__global__ __launch_bounds__(128) void proj_qk_kernel(const float* __restrict__ view,
                                                      const u16* __restrict__ Wb,
                                                      const float* __restrict__ bias,
                                                      u16* __restrict__ outb) {
  __shared__ __align__(16) u16 T[64 * 64];  // staged view^T tile [n][c] bf16
  const int tid = threadIdx.x, wave = tid >> 5, lane = tid & 31;
  const int hi = lane >> 4;
  const int b = blockIdx.z;
  const int n0 = blockIdx.x * 64;

  v8f acc[4] = {};
#pragma unroll 1
  for (int c0 = 0; c0 < CCH; c0 += 64) {
    __syncthreads();
#pragma unroll
    for (int i = 0; i < 32; ++i) {                 // 64x64 fp32 -> bf16, transpose into LDS
      const int flat = i * 128 + tid;
      const int cl = flat >> 6, nl = flat & 63;
      T[nl * 64 + cl] = f2bf(view[((size_t)b * CCH + c0 + cl) * BN + n0 + nl]);
    }
    __syncthreads();
#pragma unroll
    for (int kk = 0; kk < 2; ++kk) {
      const v16bf a = load_fragA_u16(T + wave * 16 * 64 + kk * 32, 64);
#pragma unroll
      for (int t = 0; t < 4; ++t) {
        const v16bf bfrag = load_fragB_u16(Wb + (size_t)(t * 16) * CCH + c0 + kk * 32, CCH);
        acc[t] = wmma_bf16(a, bfrag, acc[t]);
      }
    }
  }
#pragma unroll
  for (int t = 0; t < 4; ++t)
#pragma unroll
    for (int r = 0; r < 8; ++r) {
      const int o = t * 16 + (lane & 15);
      const int m = n0 + wave * 16 + r + 8 * hi;
      outb[((size_t)b * BN + m) * DQK + o] = f2bf(acc[t][r] + bias[o]);
    }
}

// ---------- pass A: v projection  out[b,o,n] = sum_c view2[b,c,n]*Wv[o,c] + bv[o] ----------
// grid (N/64, C/64, B). Wave w owns rows o = o0+16w.., 4 n-tiles. Wb = bf16 Wv row-major.
__global__ __launch_bounds__(128) void proj_v_kernel(const float* __restrict__ view,
                                                     const u16* __restrict__ Wb,
                                                     const float* __restrict__ bias,
                                                     u16* __restrict__ outb) {
  __shared__ __align__(16) u16 T[64 * 64];  // [n][c] bf16
  const int tid = threadIdx.x, wave = tid >> 5, lane = tid & 31;
  const int hi = lane >> 4;
  const int b = blockIdx.z;
  const int n0 = blockIdx.x * 64;
  const int o0 = blockIdx.y * 64;

  v8f acc[4] = {};
#pragma unroll 1
  for (int c0 = 0; c0 < CCH; c0 += 64) {
    __syncthreads();
#pragma unroll
    for (int i = 0; i < 32; ++i) {
      const int flat = i * 128 + tid;
      const int cl = flat >> 6, nl = flat & 63;
      T[nl * 64 + cl] = f2bf(view[((size_t)b * CCH + c0 + cl) * BN + n0 + nl]);
    }
    __syncthreads();
#pragma unroll
    for (int kk = 0; kk < 2; ++kk) {
      const v16bf a = load_fragA_u16(Wb + (size_t)(o0 + wave * 16) * CCH + c0 + kk * 32, CCH);
#pragma unroll
      for (int t = 0; t < 4; ++t)
        acc[t] = wmma_bf16(a, load_fragB_u16(T + t * 16 * 64 + kk * 32, 64), acc[t]);
    }
  }
#pragma unroll
  for (int t = 0; t < 4; ++t)
#pragma unroll
    for (int r = 0; r < 8; ++r) {
      const int o = o0 + wave * 16 + r + 8 * hi;
      const int n = n0 + t * 16 + (lane & 15);
      outb[((size_t)b * CCH + o) * BN + n] = f2bf(acc[t][r] + bias[o]);
    }
}

// ---------- pass B: flash attention + epilogue ----------
// grid (N/64 qtiles, C/128 cblocks, B), 128 threads (4 waves).
// Wave w: 16 query rows (q0+16w..), 128 channels. Online softmax, O = P@V accumulated in f32.
// K/V tiles double-buffered in LDS; async DMA for tile j+1 overlaps compute of tile j.
__global__ __launch_bounds__(128) void flash_kernel(const u16* __restrict__ qb,
                                                    const u16* __restrict__ kb,
                                                    const u16* __restrict__ vb,
                                                    const float* __restrict__ gamma,
                                                    const float* __restrict__ view1,
                                                    float* __restrict__ out) {
  __shared__ __align__(16) u16 Kt[2][64 * 64];     // [buf][key][d]   2 x 8 KB
  __shared__ __align__(16) u16 Vt[2][128 * 64];    // [buf][c][key]   2 x 16 KB
  __shared__ __align__(16) u16 Pt[4 * 16 * 64];    // per-wave P        8 KB
  const int tid = threadIdx.x, wave = tid >> 5, lane = tid & 31;
  const int hi = lane >> 4;
  const int b = blockIdx.z;
  const int q0 = blockIdx.x * 64;
  const int c0 = blockIdx.y * 128;

  const u16* qbase = qb + ((size_t)b * BN + q0 + wave * 16) * DQK;
  const v16bf qa0 = load_fragA_u16(qbase, DQK);        // dims 0..31
  const v16bf qa1 = load_fragA_u16(qbase + 32, DQK);   // dims 32..63

  // issue staging for tile jj into buffer jj&1
  auto stage = [&](int jj) {
    const int k0 = jj * 64;
    const u16* ksrc = kb + ((size_t)b * BN + k0) * DQK;
    const u16* vsrc = vb + ((size_t)b * CCH + c0 + tid) * (size_t)BN + k0;
    u16* kt = Kt[jj & 1];
    u16* vt = Vt[jj & 1] + tid * 64;
#if USE_ASYNC_LDS
#pragma unroll
    for (int i = 0; i < 4; ++i) {
      const int e = (i * 128 + tid) * 8;
      async_copy16(ksrc + e, kt + e);
    }
#pragma unroll
    for (int i = 0; i < 8; ++i)
      async_copy16(vsrc + i * 8, vt + i * 8);
#else
#pragma unroll
    for (int i = 0; i < 4; ++i) {
      const int e = (i * 128 + tid) * 8;
      *(uint4*)(kt + e) = *(const uint4*)(ksrc + e);
    }
#pragma unroll
    for (int i = 0; i < 8; ++i)
      ((uint4*)vt)[i] = ((const uint4*)vsrc)[i];
#endif
  };

  v8f O[8] = {};
  float mrun[8], lrun[8];
#pragma unroll
  for (int r = 0; r < 8; ++r) { mrun[r] = -1e30f; lrun[r] = 0.f; }

  stage(0);   // prologue: tile 0 in flight

#pragma unroll 1
  for (int j = 0; j < 64; ++j) {
#if USE_ASYNC_LDS
    __builtin_amdgcn_s_wait_asynccnt(0);   // tile j resident in LDS
#endif
    __syncthreads();                       // all waves done reading buffer (j+1)&1
    if (j + 1 < 64) stage(j + 1);          // DMA next tile while we compute this one
    if (j + 2 < 64)                        // nudge tile j+2 toward L2
      __builtin_prefetch(kb + ((size_t)b * BN + (j + 2) * 64) * DQK, 0, 1);

    const u16* kt = Kt[j & 1];
    const u16* vt = Vt[j & 1];

    // S = Q @ K^T  (16 rows x 64 keys per wave)
    v8f S[4];
#pragma unroll
    for (int t = 0; t < 4; ++t) {
      v8f acc = {};
      acc = wmma_bf16(qa0, load_fragB_u16(kt + t * 16 * 64, 64), acc);
      acc = wmma_bf16(qa1, load_fragB_u16(kt + t * 16 * 64 + 32, 64), acc);
      S[t] = acc;
    }

    // online softmax: row m = r + 8*hi lives on one 16-lane half
    float scale[8];
#pragma unroll
    for (int r = 0; r < 8; ++r) {
      float mx = fmaxf(fmaxf(S[0][r], S[1][r]), fmaxf(S[2][r], S[3][r]));
#pragma unroll
      for (int off = 1; off < 16; off <<= 1) mx = fmaxf(mx, __shfl_xor(mx, off, 16));
      const float nm = fmaxf(mrun[r], mx);
      scale[r] = __expf(mrun[r] - nm);
      mrun[r] = nm;
      float ls = 0.f;
#pragma unroll
      for (int t = 0; t < 4; ++t) { const float p = __expf(S[t][r] - nm); S[t][r] = p; ls += p; }
#pragma unroll
      for (int off = 1; off < 16; off <<= 1) ls += __shfl_xor(ls, off, 16);
      lrun[r] = lrun[r] * scale[r] + ls;
    }

    // P: C/D layout -> LDS row-major [m][key] (re-layout for A-operand)
    u16* P = Pt + wave * 16 * 64;
#pragma unroll
    for (int t = 0; t < 4; ++t)
#pragma unroll
      for (int r = 0; r < 8; ++r)
        P[(r + 8 * hi) * 64 + t * 16 + (lane & 15)] = f2bf(S[t][r]);

#pragma unroll
    for (int ct = 0; ct < 8; ++ct)
#pragma unroll
      for (int r = 0; r < 8; ++r) O[ct][r] *= scale[r];

    const v16bf pa0 = load_fragA_u16(P, 64);
    const v16bf pa1 = load_fragA_u16(P + 32, 64);
#pragma unroll
    for (int ct = 0; ct < 8; ++ct) {
      O[ct] = wmma_bf16(pa0, load_fragB_u16(vt + ct * 16 * 64, 64), O[ct]);
      O[ct] = wmma_bf16(pa1, load_fragB_u16(vt + ct * 16 * 64 + 32, 64), O[ct]);
    }
  }

  // epilogue: out[b,c,m] = gamma * O/l + view1
  const float g = gamma[0];
  float inv[8];
#pragma unroll
  for (int r = 0; r < 8; ++r) inv[r] = 1.0f / lrun[r];
#pragma unroll
  for (int ct = 0; ct < 8; ++ct)
#pragma unroll
    for (int r = 0; r < 8; ++r) {
      const int m = q0 + wave * 16 + r + 8 * hi;
      const int c = c0 + ct * 16 + (lane & 15);
      const size_t idx = ((size_t)b * CCH + c) * BN + m;
      out[idx] = g * (O[ct][r] * inv[r]) + view1[idx];
    }
}

// ---------- launcher ----------
extern "C" void kernel_launch(void* const* d_in, const int* in_sizes, int n_in,
                              void* d_out, int out_size, void* d_ws, size_t ws_size,
                              hipStream_t stream) {
  const float* view1 = (const float*)d_in[0];
  const float* view2 = (const float*)d_in[1];
  const float* Wq    = (const float*)d_in[2];
  const float* bq    = (const float*)d_in[3];
  const float* Wk    = (const float*)d_in[4];
  const float* bk    = (const float*)d_in[5];
  const float* Wv    = (const float*)d_in[6];
  const float* bv    = (const float*)d_in[7];
  const float* gamma = (const float*)d_in[8];
  float* out = (float*)d_out;

  // workspace: qb (4MB) | kb (4MB) | vb (32MB) | bf16 weights (~0.65MB)
  u16* qb  = (u16*)d_ws;
  u16* kb  = qb + (size_t)BATCH * BN * DQK;
  u16* vb  = kb + (size_t)BATCH * BN * DQK;
  u16* wqb = vb + (size_t)BATCH * CCH * BN;
  u16* wkb = wqb + (size_t)DQK * CCH;
  u16* wvb = wkb + (size_t)DQK * CCH;

  const int nqk = DQK * CCH, nv = CCH * CCH;
  cvt_bf16_kernel<<<(nqk + 255) / 256, 256, 0, stream>>>(Wq, wqb, nqk);
  cvt_bf16_kernel<<<(nqk + 255) / 256, 256, 0, stream>>>(Wk, wkb, nqk);
  cvt_bf16_kernel<<<(nv + 255) / 256, 256, 0, stream>>>(Wv, wvb, nv);

  const dim3 blk(128, 1, 1);
  proj_qk_kernel<<<dim3(BN / 64, 1, BATCH), blk, 0, stream>>>(view1, wqb, bq, qb);
  proj_qk_kernel<<<dim3(BN / 64, 1, BATCH), blk, 0, stream>>>(view2, wkb, bk, kb);
  proj_v_kernel<<<dim3(BN / 64, CCH / 64, BATCH), blk, 0, stream>>>(view2, wvb, bv, vb);
  flash_kernel<<<dim3(BN / 64, CCH / 128, BATCH), blk, 0, stream>>>(qb, kb, vb, gamma,
                                                                    view1, out);
}
